// Vibes_317827580315
// MI455X (gfx1250) — matrix operand order repaired and checked by reading, not compiled, not run on hardware
//
#include <hip/hip_runtime.h>
#include <hip/hip_bf16.h>

// ---------------- problem constants ----------------
#define BATCH   32
#define MAXLEN  64
#define LATENT  64
#define HIDDEN  256
#define NROWS   2048                // N = BATCH*MAXLEN
#define KDIM    4096                // MAXLEN*LATENT
#define NSTEPS  20
#define DT_F    0.05f
#define SQRT_DT 0.22360679775f
#define OUT_ELEMS (BATCH*MAXLEN*LATENT)   // 131072
#define DIAG_STRIDE 133120               // t*131072 + t*2048 : xt4[t,t,...] flat stride
#define CHUNK_K 128                      // K per LDS staging chunk
#define LDSROW  136                      // 128 + 8 pad (68 dwords) -> conflict-minimal ds_load_b128

typedef __attribute__((ext_vector_type(16))) __bf16 bf16x16;
typedef __attribute__((ext_vector_type(8)))  __bf16 bf16x8;
typedef __attribute__((ext_vector_type(8)))  float  f32x8;

// ---------------- helpers ----------------
__device__ __forceinline__ __bf16 f2bf(float f) {
  unsigned u = __builtin_bit_cast(unsigned, f);
  unsigned r = (u + 0x7FFFu + ((u >> 16) & 1u)) >> 16;   // RNE
  unsigned short s = (unsigned short)r;
  return __builtin_bit_cast(__bf16, s);
}

__device__ __forceinline__ unsigned hash_u32(unsigned x) {
  x ^= x >> 17; x *= 0xed5ad4bbu;
  x ^= x >> 11; x *= 0xac4c1b51u;
  x ^= x >> 15; x *= 0x31848babu;
  x ^= x >> 14;
  return x;
}
__device__ __forceinline__ float uniform01(unsigned h) {
  return (float)(h >> 8) * (1.0f / 16777216.0f);
}
__device__ __forceinline__ float gauss01(unsigned seed) {
  unsigned h1 = hash_u32(seed * 2u + 0x9e3779b9u);
  unsigned h2 = hash_u32(seed * 2u + 0x85ebca6bu);
  float u1 = fmaxf(uniform01(h1), 1.0f / 16777216.0f);
  float u2 = uniform01(h2);
  return sqrtf(-2.0f * logf(u1)) * cosf(6.28318530718f * u2);
}

// ---------------- prep: xt init (masked emb + uniform noise), f32 master + bf16 mirror ----------------
__global__ void k_prep_xt(const int* __restrict__ tokens, const float* __restrict__ emb,
                          float* __restrict__ xt, __bf16* __restrict__ xtb) {
  for (int idx = blockIdx.x * blockDim.x + threadIdx.x;
       idx < NROWS * KDIM; idx += gridDim.x * blockDim.x) {
    int m = idx >> 12;              // row n = p*32 + b
    int t = (idx >> 6) & 63;        // sequence position axis of xt
    int d = idx & 63;
    int p = m >> 5, b = m & 31;
    float v;
    if (p < t) v = emb[tokens[b * MAXLEN + p] * LATENT + d];
    else       v = uniform01(hash_u32((unsigned)idx ^ 0xC0FFEE01u));
    xt[idx]  = v;
    xtb[idx] = f2bf(v);
  }
}

// ---------------- prep: transpose+convert weights to bf16; zero accumulators ----------------
__global__ void k_prep_w(const float* __restrict__ W1, const float* __restrict__ W2,
                         __bf16* __restrict__ W1t, __bf16* __restrict__ W2t,
                         float* __restrict__ err, float* __restrict__ sumsq) {
  int tid = blockIdx.x * blockDim.x + threadIdx.x;
  int stride = gridDim.x * blockDim.x;
  for (int i = tid; i < HIDDEN * KDIM; i += stride) {       // W1t[n][k] = W1[k][n]
    int n = i >> 12, k = i & 4095;
    W1t[i] = f2bf(W1[k * HIDDEN + n]);
  }
  for (int i = tid; i < LATENT * HIDDEN; i += stride) {     // W2t[n][k] = W2[k][n]
    int n = i >> 8, k = i & 255;
    W2t[i] = f2bf(W2[k * LATENT + n]);
  }
  for (int i = tid; i < NROWS; i += stride) err[i] = 0.0f;
  if (tid == 0) *sumsq = 0.0f;
}

// ---------------- GEMM1: H = tanh(Xt(2048x4096) @ W1t^T + t*W1row + b1), bf16 out ----------------
// grid 128 (16-row M strips), block 256 (8 waves); wave w covers N in [32w, 32w+32).
// A tile (16 rows x 128 K) double-buffered in LDS (each thread stages one 16B load/chunk);
// 2 accumulators per wave keeps register pressure low -> accumulators stay pinned (no
// WMMA->VALU hazard nops from accumulator shuffling).
__global__ __launch_bounds__(256)
void k_gemm1_tanh(const __bf16* __restrict__ Xt, const __bf16* __restrict__ W1t,
                  const float* __restrict__ W1last, const float* __restrict__ b1,
                  float tcur, __bf16* __restrict__ H, float* __restrict__ sumsq) {
  __shared__ __bf16 As[2][16 * LDSROW];
  if (blockIdx.x == 0 && threadIdx.x == 0) *sumsq = 0.0f;   // reset norm accum for this step

  const int tid   = threadIdx.x;
  const int lane  = tid & 31;
  const int wave  = tid >> 5;            // 0..7
  const int g     = lane >> 4;           // K-half selector
  const int l16   = lane & 15;
  const int mBase = blockIdx.x << 4;     // 16 rows per block
  const int nT0   = (wave << 5) + l16;   // W1t row for N-tile 0
  const int nT1   = nT0 + 16;
  const __bf16* brow0 = W1t + (size_t)nT0 * KDIM;
  const __bf16* brow1 = W1t + (size_t)nT1 * KDIM;

  // staging map: 256 threads x 16B = 16 rows x 128 K bf16
  const int sr = tid >> 4;               // 0..15
  const int sc = (tid & 15) << 3;        // 0..120 step 8
  const __bf16* agrow = Xt + (size_t)(mBase + sr) * KDIM + sc;

  f32x8 c0 = {}, c1 = {};

  bf16x8 stage = *(const bf16x8*)(agrow);                 // chunk 0
  *(bf16x8*)(&As[0][sr * LDSROW + sc]) = stage;

  const int NCHUNK = KDIM / CHUNK_K;                      // 32
  for (int chunk = 0; chunk < NCHUNK; ++chunk) {
    __syncthreads();                                      // buf[chunk&1] ready; prev compute done
    const __bf16* as = &As[chunk & 1][0];
    if (chunk + 1 < NCHUNK)
      stage = *(const bf16x8*)(agrow + (chunk + 1) * CHUNK_K);  // prefetch next A chunk
    const int kbase = chunk * CHUNK_K;
#pragma unroll
    for (int kk = 0; kk < CHUNK_K; kk += 32) {
      bf16x8 lo = *(const bf16x8*)(as + l16 * LDSROW + kk + 16 * g);
      bf16x8 hi = *(const bf16x8*)(as + l16 * LDSROW + kk + 16 * g + 8);
      bf16x16 a;
#pragma unroll
      for (int i = 0; i < 8; ++i) { a[i] = lo[i]; a[8 + i] = hi[i]; }
      bf16x16 bf0 = *(const bf16x16*)(brow0 + kbase + kk + 16 * g);
      bf16x16 bf1 = *(const bf16x16*)(brow1 + kbase + kk + 16 * g);
      c0 = __builtin_amdgcn_wmma_f32_16x16x32_bf16(false, a, false, bf0, (short)0, c0, false, false);
      c1 = __builtin_amdgcn_wmma_f32_16x16x32_bf16(false, a, false, bf1, (short)0, c1, false, false);
    }
    if (chunk + 1 < NCHUNK)
      *(bf16x8*)(&As[(chunk + 1) & 1][sr * LDSROW + sc]) = stage; // fill other buffer
  }

  const float biasA = tcur * W1last[nT0] + b1[nT0];
  const float biasB = tcur * W1last[nT1] + b1[nT1];
#pragma unroll
  for (int e = 0; e < 8; ++e) {
    int m = mBase + e + 8 * g;
    H[(size_t)m * HIDDEN + nT0] = f2bf(tanhf(c0[e] + biasA));
    H[(size_t)m * HIDDEN + nT1] = f2bf(tanhf(c1[e] + biasB));
  }
}

// ---------------- GEMM2: score = H(2048x256) @ W2t^T + b2 ; accumulate ||score||^2 ----------------
// grid 128 (M tiles), block 128 (4 waves); wave w covers N tile [16w, 16w+16)
__global__ __launch_bounds__(128)
void k_gemm2_score(const __bf16* __restrict__ H, const __bf16* __restrict__ W2t,
                   const float* __restrict__ b2, float* __restrict__ score,
                   float* __restrict__ sumsq) {
  const int lane  = threadIdx.x & 31;
  const int wave  = threadIdx.x >> 5;
  const int mBase = blockIdx.x << 4;
  const int g     = lane >> 4;
  const int n     = (wave << 4) + (lane & 15);
  const __bf16* arow = H   + (size_t)(mBase + (lane & 15)) * HIDDEN;
  const __bf16* brow = W2t + (size_t)n * HIDDEN;
  f32x8 c = {};
  for (int k = 0; k < HIDDEN; k += 32) {
    bf16x8 lo = *(const bf16x8*)(arow + k + 16 * g);
    bf16x8 hi = *(const bf16x8*)(arow + k + 16 * g + 8);
    bf16x16 a;
#pragma unroll
    for (int i = 0; i < 8; ++i) { a[i] = lo[i]; a[8 + i] = hi[i]; }
    bf16x16 b = *(const bf16x16*)(brow + k + 16 * g);
    c = __builtin_amdgcn_wmma_f32_16x16x32_bf16(false, a, false, b, (short)0, c, false, false);
  }
  const float bias = b2[n];
  float part = 0.0f;
#pragma unroll
  for (int e = 0; e < 8; ++e) {
    int m = mBase + e + 8 * g;
    float v = c[e] + bias;
    score[(size_t)m * LATENT + n] = v;
    part += v * v;
  }
  for (int off = 16; off > 0; off >>= 1) part += __shfl_xor(part, off, 32);
  if (lane == 0) atomicAdd(sumsq, part);
}

// ---------------- err[x,y] += mean_d (labels2 - diag - score2)^2 ----------------
__global__ void k_err_acc(const float* __restrict__ xt, const int* __restrict__ tokens,
                          const float* __restrict__ emb, const float* __restrict__ score,
                          float* __restrict__ err) {
  int idx = blockIdx.x * blockDim.x + threadIdx.x;   // (x,y) flat, 0..2047
  if (idx >= NROWS) return;
  int x = idx >> 6, y = idx & 63;
  int t  = 2 * x + (y >> 5);
  int bb = y & 31;
  const float* dptr = xt + (size_t)t * DIAG_STRIDE + bb * 64;   // xt4[t,t,bb,:]
  const float* lab  = emb + (size_t)tokens[bb * MAXLEN + t] * LATENT;
  const float* sc   = score + (size_t)idx * LATENT;
  float acc = 0.0f;
  for (int d = 0; d < LATENT; ++d) {
    float df = lab[d] - dptr[d] - sc[d];
    acc += df * df;
  }
  err[idx] += acc * (1.0f / (float)LATENT);
}

// ---------------- Euler-Maruyama diagonal update: xt[n, n%64, :] += score*dt + ||score||*dW ----------------
__global__ void k_update(float* __restrict__ xt, __bf16* __restrict__ xtb,
                         const float* __restrict__ score, const float* __restrict__ sumsq,
                         int step) {
  int idx = blockIdx.x * blockDim.x + threadIdx.x;   // 0..131071 == (n,d)
  if (idx >= NROWS * LATENT) return;
  int n = idx >> 6, d = idx & 63;
  float normS = sqrtf(*sumsq);
  float dw = gauss01((unsigned)(step * (NROWS * LATENT) + idx) ^ 0xD1CEBEEFu) * SQRT_DT;
  float dx = score[idx] * DT_F + normS * dw;
  size_t pos = (size_t)n * KDIM + (size_t)(n & 63) * LATENT + d;
  float nv = xt[pos] + dx;
  xt[pos]  = nv;
  xtb[pos] = f2bf(nv);
}

// ---------------- finalize: xt_out, step_loss, sequence_loss ----------------
__global__ __launch_bounds__(256)
void k_finalize(const float* __restrict__ xt, const int* __restrict__ tokens,
                const float* __restrict__ emb, const float* __restrict__ err,
                float* __restrict__ out) {
  __shared__ float red[256];
  float absacc = 0.0f;
  for (int i = threadIdx.x; i < OUT_ELEMS; i += 256) {
    int bb = i >> 12, t = (i >> 6) & 63, d = i & 63;
    float v = xt[(size_t)t * DIAG_STRIDE + bb * 64 + d];   // transpose(xt4[t,t], (1,0,2))
    out[i] = v;
    float e = emb[(size_t)tokens[bb * MAXLEN + t] * LATENT + d];
    absacc += fabsf(e - v);
  }
  float erracc = 0.0f;
  for (int i = threadIdx.x; i < NROWS; i += 256) erracc += err[i];

  red[threadIdx.x] = absacc; __syncthreads();
  for (int s = 128; s > 0; s >>= 1) { if (threadIdx.x < s) red[threadIdx.x] += red[threadIdx.x + s]; __syncthreads(); }
  float absTot = red[0]; __syncthreads();

  red[threadIdx.x] = erracc; __syncthreads();
  for (int s = 128; s > 0; s >>= 1) { if (threadIdx.x < s) red[threadIdx.x] += red[threadIdx.x + s]; __syncthreads(); }
  if (threadIdx.x == 0) {
    out[OUT_ELEMS]     = red[0];                               // step_loss
    out[OUT_ELEMS + 1] = absTot / (float)OUT_ELEMS;            // sequence_loss
  }
}

// ---------------- launcher ----------------
extern "C" void kernel_launch(void* const* d_in, const int* in_sizes, int n_in,
                              void* d_out, int out_size, void* d_ws, size_t ws_size,
                              hipStream_t stream) {
  (void)in_sizes; (void)n_in; (void)out_size; (void)ws_size;
  const int*   tokens = (const int*)d_in[0];
  const float* emb    = (const float*)d_in[1];
  const float* W1     = (const float*)d_in[2];
  const float* b1     = (const float*)d_in[3];
  const float* W2     = (const float*)d_in[4];
  const float* b2     = (const float*)d_in[5];
  float* out = (float*)d_out;

  char* ws = (char*)d_ws;
  float*  xt    = (float*) (ws + 0);            // 2048*4096 f32  = 33554432 B
  __bf16* xtb   = (__bf16*)(ws + 33554432);     // 2048*4096 bf16 = 16777216 B
  __bf16* W1t   = (__bf16*)(ws + 50331648);     // 256*4096 bf16  =  2097152 B
  __bf16* W2t   = (__bf16*)(ws + 52428800);     // 64*256  bf16   =    32768 B
  __bf16* H     = (__bf16*)(ws + 52461568);     // 2048*256 bf16  =  1048576 B
  float*  score = (float*) (ws + 53510144);     // 2048*64 f32    =   524288 B
  float*  err   = (float*) (ws + 54034432);     // 2048 f32       =     8192 B
  float*  sumsq = (float*) (ws + 54042624);     // 1 f32

  k_prep_xt<<<512, 256, 0, stream>>>(tokens, emb, xt, xtb);
  k_prep_w <<<256, 256, 0, stream>>>(W1, W2, W1t, W2t, err, sumsq);

  const float* W1last = W1 + (size_t)KDIM * HIDDEN;   // row 4096 of W1 (time column)
  for (int s = 0; s < NSTEPS; ++s) {
    float tcur = DT_F * (float)s;
    k_gemm1_tanh <<<128, 256, 0, stream>>>(xtb, W1t, W1last, b1, tcur, H, sumsq);
    k_gemm2_score<<<128, 128, 0, stream>>>(H, W2t, b2, score, sumsq);
    k_err_acc    <<<8,   256, 0, stream>>>(xt, tokens, emb, score, err);
    k_update     <<<512, 256, 0, stream>>>(xt, xtb, score, sumsq, s);
  }

  k_finalize<<<1, 256, 0, stream>>>(xt, tokens, emb, err, out);
}
